// EMMPTNet_unbalance_34368328302741
// MI455X (gfx1250) — matrix operand order repaired and checked by reading, not compiled, not run on hardware
//
#include <hip/hip_runtime.h>
#include <math.h>

// ---------------------------------------------------------------------------
// EMMPTNet forward for MI455X (gfx1250, wave32, WMMA, async-to-LDS).
//  - All K>=128 GEMMs: v_wmma_f32_16x16x32_f16, f16 operands pre-converted
//    once in scratch, fp32 accumulate, 1x4 N-tile register blocking.
//  - LSTM recurrence: 256KB w_hh transposed-resident in 320KB LDS.
//  - GCN scatter: global_atomic_add_f32 (HBM-bound hotspot).
//  - Broadcast-heavy reads staged to LDS via global_load_async_to_lds_b32.
// ---------------------------------------------------------------------------

typedef __attribute__((ext_vector_type(16))) _Float16 v16h;
typedef __attribute__((ext_vector_type(8)))  _Float16 v8h;
typedef __attribute__((ext_vector_type(8)))  float    v8f;

#define T_SEQ 2048
#define HDIM  128
#define DIN_  17
#define NNODE 50000
#define NEDGE 600000

__device__ __forceinline__ float sigm(float x) { return 1.0f / (1.0f + __expf(-x)); }

// ---- CDNA5 async copy global -> LDS (ASYNCcnt), drained by s_wait_asynccnt --
__device__ __forceinline__ void async_b32(void* lds_dst, const float* g_src)
{
    uint32_t l = (uint32_t)(uintptr_t)lds_dst;   // low 32 bits of generic LDS ptr = LDS offset
    asm volatile("global_load_async_to_lds_b32 %0, %1, off"
                 :: "v"(l), "v"(g_src) : "memory");
}
__device__ __forceinline__ void async_wait0()
{
    asm volatile("s_wait_asynccnt 0x0" ::: "memory");
}

// ---------------------------------------------------------------------------
// WMMA GEMM, NT form: C(MxN) = scale * A(MxK) @ B(NxK)^T + bias1[n] + bias2[n]
// A, B row-major _Float16; one wave computes a 16x64 strip (4 N-tiles) so each
// A fragment feeds 4 v_wmma ops. M%16==0, N%64==0, K%32==0 (all shapes comply).
// Fragment layouts: A and B both contiguous K (k = 16*(lane/16) + s).
// ---------------------------------------------------------------------------
template <bool RELU, bool F16OUT>
__global__ void wmma_gemm_nt(const _Float16* __restrict__ A, const _Float16* __restrict__ B,
                             const float* __restrict__ bias1, const float* __restrict__ bias2,
                             void* __restrict__ Cv, int M, int N, int K,
                             int lda, int ldb, int ldc, float scale)
{
    const int wid  = (blockIdx.x * blockDim.x + threadIdx.x) >> 5;
    const int lane = threadIdx.x & 31;
    const int ngrp = N >> 6;                    // groups of four 16-wide N tiles
    const int total = (M >> 4) * ngrp;
    if (wid >= total) return;                   // wave-uniform: EXEC all-1s for WMMA
    const int tm  = wid / ngrp;
    const int tn0 = (wid - tm * ngrp) << 2;     // first of 4 N tiles

    const int r   = lane & 15;
    const int kbA = (lane >> 4) << 4;           // A: contiguous K run 0 / 16
    const int kbB = (lane >> 4) << 4;           // B: contiguous K run 0 / 16

    const _Float16* Arow = A + (size_t)(tm * 16 + r) * lda + kbA;
    const _Float16* B0   = B + (size_t)((tn0 + 0) * 16 + r) * ldb + kbB;
    const _Float16* B1   = B + (size_t)((tn0 + 1) * 16 + r) * ldb + kbB;
    const _Float16* B2   = B + (size_t)((tn0 + 2) * 16 + r) * ldb + kbB;
    const _Float16* B3   = B + (size_t)((tn0 + 3) * 16 + r) * ldb + kbB;

    v8f acc0 = {}, acc1 = {}, acc2 = {}, acc3 = {};
    for (int k0 = 0; k0 < K; k0 += 32) {
        v8h alo = *(const v8h*)(Arow + k0);            // global_load_b128
        v8h ahi = *(const v8h*)(Arow + k0 + 8);
        v16h a = __builtin_shufflevector(alo, ahi, 0, 1, 2, 3, 4, 5, 6, 7,
                                         8, 9, 10, 11, 12, 13, 14, 15);
        v16h b0 = *(const v16h*)(B0 + k0);
        v16h b1 = *(const v16h*)(B1 + k0);
        v16h b2 = *(const v16h*)(B2 + k0);
        v16h b3 = *(const v16h*)(B3 + k0);
        acc0 = __builtin_amdgcn_wmma_f32_16x16x32_f16(false, a, false, b0, (short)0, acc0, false, false);
        acc1 = __builtin_amdgcn_wmma_f32_16x16x32_f16(false, a, false, b1, (short)0, acc1, false, false);
        acc2 = __builtin_amdgcn_wmma_f32_16x16x32_f16(false, a, false, b2, (short)0, acc2, false, false);
        acc3 = __builtin_amdgcn_wmma_f32_16x16x32_f16(false, a, false, b3, (short)0, acc3, false, false);
    }

    const int mb = tm * 16 + ((lane >> 4) << 3);       // C layout: n = lane%16, m = mb+vgpr
    v8f accs[4] = {acc0, acc1, acc2, acc3};
#pragma unroll
    for (int tt = 0; tt < 4; ++tt) {
        const int n = (tn0 + tt) * 16 + (lane & 15);
        float bvv = 0.0f;
        if (bias1) bvv += bias1[n];
        if (bias2) bvv += bias2[n];
#pragma unroll
        for (int rr = 0; rr < 8; ++rr) {
            float val = accs[tt][rr] * scale + bvv;
            if (RELU) val = fmaxf(val, 0.0f);
            if (F16OUT) ((_Float16*)Cv)[(size_t)(mb + rr) * ldc + n] = (_Float16)val;
            else        ((float*)Cv)   [(size_t)(mb + rr) * ldc + n] = val;
        }
    }
}

// ---------------------------------------------------------------------------
// f32 -> f16 conversions (one-time pre-passes feeding the WMMA pipes).
// ---------------------------------------------------------------------------
__global__ void cvt_f16(const float* __restrict__ src, _Float16* __restrict__ dst, int n)
{
    int i = blockIdx.x * blockDim.x + threadIdx.x;
    if (i < n) dst[i] = (_Float16)src[i];
}

// src [R][C] f32 -> dst [C][R] f16 (weight transpose so every GEMM is NT form)
__global__ void cvt_f16_transpose(const float* __restrict__ src, _Float16* __restrict__ dst,
                                  int R, int C)
{
    int i = blockIdx.x * blockDim.x + threadIdx.x;
    if (i >= R * C) return;
    int rr = i / C, cc = i - rr * C;
    dst[(size_t)cc * R + rr] = (_Float16)src[i];
}

// ---------------------------------------------------------------------------
// Layer-0 input projection (K=17, VALU): xg[d][t][g] = x . w_ih0 + b_ih + b_hh
// ---------------------------------------------------------------------------
__global__ void xg0_kernel(const float* __restrict__ x, const float* __restrict__ w,
                           const float* __restrict__ bi, const float* __restrict__ bh,
                           float* __restrict__ xg)
{
    int idx = blockIdx.x * blockDim.x + threadIdx.x;
    if (idx >= 2 * T_SEQ * 512) return;
    int d   = idx / (T_SEQ * 512);
    int rem = idx - d * (T_SEQ * 512);
    int t   = rem >> 9;
    int g   = rem & 511;
    const float* wr = w + (size_t)(d * 512 + g) * DIN_;
    const float* xr = x + (size_t)t * DIN_;
    float s = bi[d * 512 + g] + bh[d * 512 + g];
#pragma unroll
    for (int k = 0; k < DIN_; ++k) s += xr[k] * wr[k];
    xg[idx] = s;   // layout [d][t][g]
}

// ---------------------------------------------------------------------------
// Bidirectional LSTM scan. blockIdx.x = direction. 512 threads (16 waves).
// w_hh (256KB) transposed-resident in LDS (consecutive lanes -> consecutive
// banks, conflict-free). hs written directly as f16 (sole consumer is WMMA-A).
// ---------------------------------------------------------------------------
__global__ void lstm_scan(const float* __restrict__ xg,   // [2, T, 512]
                          const float* __restrict__ w_hh, // [2, 512, 128]
                          _Float16* __restrict__ hs)      // [T, 256] f16
{
    extern __shared__ float smem[];
    float* w_lds = smem;              // [128][512] transposed
    float* h_lds = smem + 65536;      // 128
    float* c_lds = smem + 65664;      // 128
    float* gbuf  = smem + 65792;      // 512

    const int d = blockIdx.x;
    const int n = threadIdx.x;        // gate row 0..511

    for (int i = n; i < 512 * 128; i += 512) {
        int row = i >> 7, k = i & 127;
        w_lds[k * 512 + row] = w_hh[(size_t)(d * 512 + row) * 128 + k];
    }
    if (n < 128) { h_lds[n] = 0.0f; c_lds[n] = 0.0f; }
    __syncthreads();

    for (int s = 0; s < T_SEQ; ++s) {
        const int t = d ? (T_SEQ - 1 - s) : s;
        float g = xg[((size_t)d * T_SEQ + t) * 512 + n];
        if (s + 1 < T_SEQ) {          // hide the only global latency in the chain
            const int tn = d ? (T_SEQ - 2 - s) : s + 1;
            __builtin_prefetch(xg + ((size_t)d * T_SEQ + tn) * 512 + n, 0, 0);
        }
#pragma unroll 4
        for (int k = 0; k < 128; ++k)
            g = fmaf(h_lds[k], w_lds[k * 512 + n], g);
        gbuf[n] = g;
        __syncthreads();
        if (n < 128) {
            float ig = sigm(gbuf[n]);
            float fg = sigm(gbuf[128 + n]);
            float gg = tanhf(gbuf[256 + n]);
            float og = sigm(gbuf[384 + n]);
            float c  = fg * c_lds[n] + ig * gg;
            float h  = og * tanhf(c);
            c_lds[n] = c;
            h_lds[n] = h;
            hs[(size_t)t * 256 + d * 128 + n] = (_Float16)h;
        }
        __syncthreads();
    }
}

// ---------------------------------------------------------------------------
// Row softmax in place (one block per row).
// ---------------------------------------------------------------------------
__global__ void softmax_rows(float* __restrict__ x, int cols)
{
    __shared__ float red[256];
    float* p = x + (size_t)blockIdx.x * cols;
    int tid = threadIdx.x;

    float m = -3.4e38f;
    for (int i = tid; i < cols; i += 256) m = fmaxf(m, p[i]);
    red[tid] = m; __syncthreads();
    for (int s = 128; s > 0; s >>= 1) { if (tid < s) red[tid] = fmaxf(red[tid], red[tid + s]); __syncthreads(); }
    m = red[0]; __syncthreads();

    float sum = 0.0f;
    for (int i = tid; i < cols; i += 256) sum += __expf(p[i] - m);
    red[tid] = sum; __syncthreads();
    for (int s = 128; s > 0; s >>= 1) { if (tid < s) red[tid] += red[tid + s]; __syncthreads(); }
    float inv = 1.0f / red[0];

    for (int i = tid; i < cols; i += 256) p[i] = __expf(p[i] - m) * inv;
}

// ---------------------------------------------------------------------------
// ctx = attn_w[T-1,:] @ v ; gmol = ctx @ Wfc^T + bfc.
// The attention row is broadcast-read 128x -> stage it in LDS with the CDNA5
// async copy path (ASYNCcnt + s_wait_asynccnt).
// ---------------------------------------------------------------------------
__global__ void ctx_gmol_kernel(const float* __restrict__ attn_last,
                                const float* __restrict__ v,
                                const float* __restrict__ Wfc, const float* __restrict__ bfc,
                                float* __restrict__ gmol)
{
    __shared__ float arow[T_SEQ];
    __shared__ float ctx[HDIM];
    int tid = threadIdx.x;

    for (int i = tid; i < T_SEQ; i += 256)
        async_b32(&arow[i], attn_last + i);
    async_wait0();
    __syncthreads();

    if (tid < HDIM) {
        float s = 0.0f;
        for (int t = 0; t < T_SEQ; ++t) s = fmaf(arow[t], v[(size_t)t * HDIM + tid], s);
        ctx[tid] = s;
    }
    __syncthreads();
    for (int n = tid; n < 256; n += blockDim.x) {
        float s = bfc[n];
        const float* wr = Wfc + (size_t)n * HDIM;
        for (int j = 0; j < HDIM; ++j) s = fmaf(ctx[j], wr[j], s);
        gmol[n] = s;
    }
}

// ---------------------------------------------------------------------------
// GCN edge scatter: agg[dst] += feat[src] * ew (4 floats/thread, f32 atomics)
// ---------------------------------------------------------------------------
__global__ void edge_scatter(const float* __restrict__ feat, const int* __restrict__ src,
                             const int* __restrict__ dst, const float* __restrict__ ew,
                             float* __restrict__ agg, int F)
{
    int idx = blockIdx.x * blockDim.x + threadIdx.x;
    int per = F >> 2;
    if (idx >= NEDGE * per) return;
    int e = idx / per;
    int f = (idx - e * per) << 2;
    float w = ew[e];
    const float* fp = feat + (size_t)src[e] * F + f;
    float*       ap = agg  + (size_t)dst[e] * F + f;
    atomicAdd(ap + 0, fp[0] * w);
    atomicAdd(ap + 1, fp[1] * w);
    atomicAdd(ap + 2, fp[2] * w);
    atomicAdd(ap + 3, fp[3] * w);
}

// ---------------------------------------------------------------------------
// Column mean over rows: one block per column.
// ---------------------------------------------------------------------------
__global__ void col_mean(const float* __restrict__ h2, float* __restrict__ hg, int rows, int cols)
{
    __shared__ float red[256];
    int j = blockIdx.x, tid = threadIdx.x;
    float s = 0.0f;
    for (int i = tid; i < rows; i += 256) s += h2[(size_t)i * cols + j];
    red[tid] = s; __syncthreads();
    for (int st = 128; st > 0; st >>= 1) { if (tid < st) red[tid] += red[tid + st]; __syncthreads(); }
    if (tid == 0) hg[j] = red[0] / (float)rows;
}

// ---------------------------------------------------------------------------
// feats = concat(hg[256], gmol[256], smiles[574], kmer[64]) -> 1150
// ---------------------------------------------------------------------------
__global__ void concat_feats(const float* __restrict__ hg, const float* __restrict__ gmol,
                             const float* __restrict__ smiles, const float* __restrict__ kmer,
                             float* __restrict__ feats)
{
    int i = blockIdx.x * blockDim.x + threadIdx.x;
    if (i >= 1150) return;
    float v;
    if (i < 256)       v = hg[i];
    else if (i < 512)  v = gmol[i - 256];
    else if (i < 1086) v = smiles[i - 512];
    else               v = kmer[i - 1086];
    feats[i] = v;
}

// ---------------------------------------------------------------------------
// Fused MLP head (1150->575->256->64->1); feats staged to LDS via async copy.
// ---------------------------------------------------------------------------
__global__ void mlp_head(const float* __restrict__ feats_g,
                         const float* __restrict__ W1, const float* __restrict__ b1,
                         const float* __restrict__ W2, const float* __restrict__ b2,
                         const float* __restrict__ W3, const float* __restrict__ b3,
                         const float* __restrict__ W4, const float* __restrict__ b4,
                         float* __restrict__ out)
{
    __shared__ float f[1150];
    __shared__ float a1[575];
    __shared__ float a2[256];
    __shared__ float a3[64];
    int tid = threadIdx.x, bd = blockDim.x;

    for (int i = tid; i < 1150; i += bd)
        async_b32(&f[i], feats_g + i);
    async_wait0();
    __syncthreads();

    for (int n = tid; n < 575; n += bd) {
        float s = b1[n];
        const float* wr = W1 + (size_t)n * 1150;
        for (int k = 0; k < 1150; ++k) s = fmaf(f[k], wr[k], s);
        a1[n] = fmaxf(s, 0.0f);
    }
    __syncthreads();
    for (int n = tid; n < 256; n += bd) {
        float s = b2[n];
        const float* wr = W2 + (size_t)n * 575;
        for (int k = 0; k < 575; ++k) s = fmaf(a1[k], wr[k], s);
        a2[n] = fmaxf(s, 0.0f);
    }
    __syncthreads();
    for (int n = tid; n < 64; n += bd) {
        float s = b3[n];
        const float* wr = W3 + (size_t)n * 256;
        for (int k = 0; k < 256; ++k) s = fmaf(a2[k], wr[k], s);
        a3[n] = fmaxf(s, 0.0f);
    }
    __syncthreads();
    if (tid == 0) {
        float s = b4[0];
        for (int k = 0; k < 64; ++k) s = fmaf(a3[k], W4[k], s);
        out[0] = s;
    }
}

// ---------------------------------------------------------------------------
// Host-side orchestration.
// ---------------------------------------------------------------------------
static inline int ceil_div(long long a, long long b) { return (int)((a + b - 1) / b); }

extern "C" void kernel_launch(void* const* d_in, const int* in_sizes, int n_in,
                              void* d_out, int out_size, void* d_ws, size_t ws_size,
                              hipStream_t stream)
{
    const float* g_mol   = (const float*)d_in[0];
    const float* feat_sq = (const float*)d_in[1];
    const float* smiles  = (const float*)d_in[2];
    const float* kmer    = (const float*)d_in[3];
    const float* edge_w  = (const float*)d_in[4];
    const float* w_ih0   = (const float*)d_in[5];
    const float* w_hh0   = (const float*)d_in[6];
    const float* b_ih0   = (const float*)d_in[7];
    const float* b_hh0   = (const float*)d_in[8];
    const float* w_ih1   = (const float*)d_in[9];
    const float* w_hh1   = (const float*)d_in[10];
    const float* b_ih1   = (const float*)d_in[11];
    const float* b_hh1   = (const float*)d_in[12];
    const float* Wq      = (const float*)d_in[13];
    const float* bq      = (const float*)d_in[14];
    const float* Wk      = (const float*)d_in[15];
    const float* bk      = (const float*)d_in[16];
    const float* Wv      = (const float*)d_in[17];
    const float* bv      = (const float*)d_in[18];
    const float* Wfc     = (const float*)d_in[19];
    const float* bfc     = (const float*)d_in[20];
    const float* W_gc1   = (const float*)d_in[21];
    const float* b_gc1   = (const float*)d_in[22];
    const float* W_gc2   = (const float*)d_in[23];
    const float* b_gc2   = (const float*)d_in[24];
    const float* W1      = (const float*)d_in[25];
    const float* b1      = (const float*)d_in[26];
    const float* W2      = (const float*)d_in[27];
    const float* b2      = (const float*)d_in[28];
    const float* W3      = (const float*)d_in[29];
    const float* b3      = (const float*)d_in[30];
    const float* W4      = (const float*)d_in[31];
    const float* b4      = (const float*)d_in[32];
    const int*   e_src   = (const int*)d_in[33];
    const int*   e_dst   = (const int*)d_in[34];

    float* out  = (float*)d_out;
    float* attn = out + 1;                       // [T, T] attention output

    // ---- workspace layout: f32 region, then 32B-aligned f16 region --------
    float* ws  = (float*)d_ws;
    size_t off = 0;
    float* xg   = ws + off; off += (size_t)2 * T_SEQ * 512;   // reused layer0/layer1
    float* vmat = ws + off; off += (size_t)T_SEQ * HDIM;
    float* agg  = ws + off; off += (size_t)NNODE * HDIM;      // reused agg1 -> agg2
    float* h1g  = ws + off; off += (size_t)NNODE * HDIM;
    float* h2   = ws + off; off += (size_t)NNODE * 256;
    float* gmol = ws + off; off += 256;
    float* hg   = ws + off; off += 256;
    float* feats= ws + off; off += 1152;
    off = (off + 15) & ~(size_t)15;              // 64B-align the half region

    _Float16* hbase = (_Float16*)(ws + off);
    size_t ho = 0;
    _Float16* h0h    = hbase + ho; ho += (size_t)T_SEQ * 256;
    _Float16* h1h    = hbase + ho; ho += (size_t)T_SEQ * 256;
    _Float16* wih1h  = hbase + ho; ho += (size_t)2 * 512 * 256;
    _Float16* Wqh    = hbase + ho; ho += (size_t)HDIM * 256;
    _Float16* Wkh    = hbase + ho; ho += (size_t)HDIM * 256;
    _Float16* Wvh    = hbase + ho; ho += (size_t)HDIM * 256;
    _Float16* qh     = hbase + ho; ho += (size_t)T_SEQ * HDIM;
    _Float16* kh     = hbase + ho; ho += (size_t)T_SEQ * HDIM;
    _Float16* aggh   = hbase + ho; ho += (size_t)NNODE * HDIM;
    _Float16* Wg1Th  = hbase + ho; ho += (size_t)HDIM * HDIM;   // W_gc1^T
    _Float16* Wg2Th  = hbase + ho; ho += (size_t)256 * HDIM;    // W_gc2^T

    const int lstm_lds = (512 * 128 + 128 + 128 + 512) * (int)sizeof(float); // 259KB

    // ---- one-time weight conversions (overlappable, tiny) ----
    cvt_f16<<<ceil_div(2 * 512 * 256, 256), 256, 0, stream>>>(w_ih1, wih1h, 2 * 512 * 256);
    cvt_f16<<<ceil_div(HDIM * 256, 256), 256, 0, stream>>>(Wq, Wqh, HDIM * 256);
    cvt_f16<<<ceil_div(HDIM * 256, 256), 256, 0, stream>>>(Wk, Wkh, HDIM * 256);
    cvt_f16<<<ceil_div(HDIM * 256, 256), 256, 0, stream>>>(Wv, Wvh, HDIM * 256);
    cvt_f16_transpose<<<ceil_div(HDIM * HDIM, 256), 256, 0, stream>>>(W_gc1, Wg1Th, HDIM, HDIM);
    cvt_f16_transpose<<<ceil_div(HDIM * 256, 256), 256, 0, stream>>>(W_gc2, Wg2Th, HDIM, 256);

    // ---- LSTM layer 0 ----
    xg0_kernel<<<ceil_div((long long)2 * T_SEQ * 512, 256), 256, 0, stream>>>(
        g_mol, w_ih0, b_ih0, b_hh0, xg);
    lstm_scan<<<2, 512, lstm_lds, stream>>>(xg, w_hh0, h0h);

    // ---- LSTM layer 1 input projection: xg1[d] = h0 @ w_ih1[d]^T + biases ----
    for (int d = 0; d < 2; ++d) {
        int waves = (T_SEQ / 16) * (512 / 64);
        wmma_gemm_nt<false, false><<<ceil_div((long long)waves * 32, 256), 256, 0, stream>>>(
            h0h, wih1h + (size_t)d * 512 * 256, b_ih1 + d * 512, b_hh1 + d * 512,
            xg + (size_t)d * T_SEQ * 512, T_SEQ, 512, 256, 256, 256, 512, 1.0f);
    }
    lstm_scan<<<2, 512, lstm_lds, stream>>>(xg, w_hh1, h1h);

    // ---- attention projections: q/k straight to f16, v stays f32 ----
    {
        int waves = (T_SEQ / 16) * (HDIM / 64);
        int blocks = ceil_div((long long)waves * 32, 256);
        wmma_gemm_nt<false, true><<<blocks, 256, 0, stream>>>(
            h1h, Wqh, bq, nullptr, qh, T_SEQ, HDIM, 256, 256, 256, HDIM, 1.0f);
        wmma_gemm_nt<false, true><<<blocks, 256, 0, stream>>>(
            h1h, Wkh, bk, nullptr, kh, T_SEQ, HDIM, 256, 256, 256, HDIM, 1.0f);
        wmma_gemm_nt<false, false><<<blocks, 256, 0, stream>>>(
            h1h, Wvh, bv, nullptr, vmat, T_SEQ, HDIM, 256, 256, 256, HDIM, 1.0f);
    }

    // ---- scores = q @ k^T / sqrt(H), written straight into d_out ----
    {
        int waves = (T_SEQ / 16) * (T_SEQ / 64);
        wmma_gemm_nt<false, false><<<ceil_div((long long)waves * 32, 256), 256, 0, stream>>>(
            qh, kh, nullptr, nullptr, attn, T_SEQ, T_SEQ, HDIM, HDIM, HDIM, T_SEQ,
            0.08838834764831845f /* 1/sqrt(128) */);
    }
    softmax_rows<<<T_SEQ, 256, 0, stream>>>(attn, T_SEQ);

    // ---- gmol from the last attention row only ----
    ctx_gmol_kernel<<<1, 256, 0, stream>>>(
        attn + (size_t)(T_SEQ - 1) * T_SEQ, vmat, Wfc, bfc, gmol);

    // ---- GCN layer 1 ----
    hipMemsetAsync(agg, 0, (size_t)NNODE * HDIM * sizeof(float), stream);
    edge_scatter<<<ceil_div((long long)NEDGE * (HDIM / 4), 256), 256, 0, stream>>>(
        feat_sq, e_src, e_dst, edge_w, agg, HDIM);
    cvt_f16<<<ceil_div(NNODE * HDIM, 256), 256, 0, stream>>>(agg, aggh, NNODE * HDIM);
    {
        int waves = (NNODE / 16) * (HDIM / 64);
        wmma_gemm_nt<true, false><<<ceil_div((long long)waves * 32, 256), 256, 0, stream>>>(
            aggh, Wg1Th, b_gc1, nullptr, h1g, NNODE, HDIM, HDIM, HDIM, HDIM, HDIM, 1.0f);
    }

    // ---- GCN layer 2 ----
    hipMemsetAsync(agg, 0, (size_t)NNODE * HDIM * sizeof(float), stream);
    edge_scatter<<<ceil_div((long long)NEDGE * (HDIM / 4), 256), 256, 0, stream>>>(
        h1g, e_src, e_dst, edge_w, agg, HDIM);
    cvt_f16<<<ceil_div(NNODE * HDIM, 256), 256, 0, stream>>>(agg, aggh, NNODE * HDIM);
    {
        int waves = (NNODE / 16) * (256 / 64);
        wmma_gemm_nt<false, false><<<ceil_div((long long)waves * 32, 256), 256, 0, stream>>>(
            aggh, Wg2Th, b_gc2, nullptr, h2, NNODE, 256, HDIM, HDIM, HDIM, 256, 1.0f);
    }
    col_mean<<<256, 256, 0, stream>>>(h2, hg, NNODE, 256);

    // ---- head ----
    concat_feats<<<ceil_div(1150, 256), 256, 0, stream>>>(hg, gmol, smiles, kmer, feats);
    mlp_head<<<1, 640, 0, stream>>>(feats, W1, b1, W2, b2, W3, b3, W4, b4, out);
}
